// PNLM_22771916604107
// MI455X (gfx1250) — matrix operand order repaired and checked by reading, not compiled, not run on hardware
//
#include <hip/hip_runtime.h>
#include <hip/hip_bf16.h>
#include <math.h>

// ---- types -----------------------------------------------------------------
typedef __bf16 bf16_t;
typedef bf16_t v16bf __attribute__((ext_vector_type(16)));
typedef float  v8f   __attribute__((ext_vector_type(8)));
typedef unsigned int u32x4 __attribute__((ext_vector_type(4)));

union ABPack { u32x4 q[2]; v16bf v; };
union DPack  { u32x4 q;    bf16_t h[8]; };

// ---- WMMA helpers (CDNA5 v_wmma_f32_16x16x32_bf16) -------------------------
__device__ __forceinline__ v8f wmma_bf16(v16bf a, v16bf b, v8f c) {
  // 8 args: (neg_a, A, neg_b, B, c_mod, C, reuse_a, reuse_b)
  return __builtin_amdgcn_wmma_f32_16x16x32_bf16(false, a, false, b,
                                                 (short)0, c, false, false);
}

// A tile 16(M) x 32(K) bf16 from row-major source (stride lda elements).
// Per ISA 7.12.2: lane(r,half): v0-3 = K = half*8+0..7 ; v4-7 = K = 16+half*8+0..7
template <typename PT>
__device__ __forceinline__ v16bf load_tile_A(const PT* src, int lda, int lane) {
  int r = lane & 15, half = lane >> 4;
  const bf16_t* base = (const bf16_t*)src + (size_t)r * lda + half * 8;
  ABPack p;
  p.q[0] = *(const u32x4*)(base);        // K = half*8 .. +7
  p.q[1] = *(const u32x4*)(base + 16);   // K = 16+half*8 .. +7
  return p.v;
}

// B tile 32(K) x 16(N) bf16 from row-major source (stride ldb elements).
// lane(r,half) holds column N = r; elem s: K = half*8+s (s<8), 16+half*8+(s-8) (s>=8).
__device__ __forceinline__ v16bf load_tile_B(const bf16_t* src, int ldb, int lane) {
  int r = lane & 15, half = lane >> 4;
  v16bf v;
  #pragma unroll
  for (int s = 0; s < 16; ++s) {
    int k = (s < 8) ? (half * 8 + s) : (16 + half * 8 + (s - 8));
    v[s] = src[(size_t)k * ldb + r];
  }
  return v;
}

// D layout: lane(r,half), vgpr g -> M = g + half*8, N = r.

// ---- weight prep: fp32 -> padded bf16 in d_ws ------------------------------
// ws layout: wq_bf [160][256] | wk_bf [160][256] | wp_bf [256][256]
__global__ void pnlm_prep_weights(const float* __restrict__ wq,
                                  const float* __restrict__ wk,
                                  const float* __restrict__ wp,
                                  bf16_t* __restrict__ ws) {
  int idx = blockIdx.x * 256 + threadIdx.x;
  if (idx < 40960) {                       // wq padded 144->160 rows
    int e = idx >> 8, c = idx & 255;
    ws[idx] = (e < 144) ? (bf16_t)wq[e * 256 + c] : (bf16_t)0.0f;
  } else if (idx < 81920) {                // wk padded
    int i = idx - 40960; int e = i >> 8, c = i & 255;
    ws[idx] = (e < 144) ? (bf16_t)wk[e * 256 + c] : (bf16_t)0.0f;
  } else if (idx < 147456) {               // wproj as-is
    ws[idx] = (bf16_t)wp[idx - 81920];
  }
}

// ---- main kernel: one workgroup per patch-block ----------------------------
// LDS map (bytes):
//   Xs   bf16[256][256]   @0        131072   (cols 225..255 zero)
//   Qs   bf16[240][160]   @131072    76800   (n-major)
//   Km   bf16[160][240]   @207872    76800   (e-major)
//   attF f32 [16][240]    @284672    15360
//   attT bf16[256][16]    @300032     8192   (m-major, rows>=225 zero)
//   yB   bf16[256][16]    @308224     8192   (c-major)
//   redS f32 [16][16]     @316416     1024   (softmax reduction scratch)
// total 317440 <= 320KB
#define LDS_BYTES 317440

__launch_bounds__(256)
__global__ void pnlm_main(const float* __restrict__ x,
                          const bf16_t* __restrict__ wqb,
                          const bf16_t* __restrict__ wkb,
                          const bf16_t* __restrict__ wpb,
                          float* __restrict__ out) {
  extern __shared__ char smem[];
  bf16_t* Xs   = (bf16_t*)(smem);
  bf16_t* Qs   = (bf16_t*)(smem + 131072);
  bf16_t* Km   = (bf16_t*)(smem + 207872);
  float*  attF = (float*) (smem + 284672);
  bf16_t* attT = (bf16_t*)(smem + 300032);
  bf16_t* yB   = (bf16_t*)(smem + 308224);
  float*  redS = (float*) (smem + 316416);

  const int tid  = threadIdx.x;
  const int lane = tid & 31;
  const int wave = tid >> 5;
  const int r    = lane & 15;
  const int half = lane >> 4;

  const int blk = blockIdx.x;          // 2048 = 8 batch * 256 patches
  const int bi  = blk >> 8;
  const int t   = blk & 255;
  const int h0  = (t >> 4) * 15;
  const int w0  = (t & 15) * 15;

  // ---- load X patch -> bf16 LDS (c-major, 256-wide rows) -------------------
  for (int s = tid; s < 256 * 15; s += 256) {
    int c = s / 15, row = s % 15;
    const float* src = x + ((size_t)(bi * 256 + c) * 240 + (h0 + row)) * 240 + w0;
    bf16_t* dst = Xs + c * 256 + row * 15;
    #pragma unroll
    for (int j = 0; j < 15; ++j) dst[j] = (bf16_t)src[j];
  }
  for (int i = tid; i < 256 * 31; i += 256) {        // zero pad cols 225..255
    int c = i / 31, j = i % 31;
    Xs[c * 256 + 225 + j] = (bf16_t)0.0f;
  }
  for (int i = tid; i < 31 * 16; i += 256)           // zero attT pad rows
    attT[225 * 16 + i] = (bf16_t)0.0f;
  __syncthreads();

  // ---- Q^T = Wq*X, stored transposed -> Qs[n][e];  Km = Wk*X -> Km[e][m] ---
  for (int job = wave; job < 150; job += 8) {        // 15 n-tiles x 10 e-tiles
    int nt = job / 10, et = job % 10;
    v8f acc = {};
    #pragma unroll
    for (int kc = 0; kc < 8; ++kc) {
      v16bf a = load_tile_A(wqb + (size_t)et * 16 * 256 + kc * 32, 256, lane);
      v16bf b = load_tile_B(Xs + kc * 32 * 256 + nt * 16, 256, lane);
      acc = wmma_bf16(a, b, acc);
    }
    DPack d;                                         // transpose on writeback
    #pragma unroll
    for (int g = 0; g < 8; ++g) d.h[g] = (bf16_t)acc[g];
    *(u32x4*)(Qs + (nt * 16 + r) * 160 + et * 16 + half * 8) = d.q;
  }
  for (int job = wave; job < 150; job += 8) {
    int nt = job / 10, et = job % 10;
    v8f acc = {};
    #pragma unroll
    for (int kc = 0; kc < 8; ++kc) {
      v16bf a = load_tile_A(wkb + (size_t)et * 16 * 256 + kc * 32, 256, lane);
      v16bf b = load_tile_B(Xs + kc * 32 * 256 + nt * 16, 256, lane);
      acc = wmma_bf16(a, b, acc);
    }
    #pragma unroll
    for (int g = 0; g < 8; ++g)                      // straight store e-major
      Km[(et * 16 + half * 8 + g) * 240 + nt * 16 + r] = (bf16_t)acc[g];
  }
  __syncthreads();

  const float sm_scale = (1.0f / 12.0f) * 1.44269504f;   // 1/sqrt(144) * log2(e)
  const int srow = tid / 15;                             // softmax row (0..15)
  const int sseg = tid % 15;                             // 15-col segment

  // ---- per 16-row n-tile: att -> softmax -> Y -> Wproj*Y + x ---------------
  for (int nt = 0; nt < 15; ++nt) {
    // att row-tile (16 x 240): A from Qs, B from Km; 2 m-tiles per wave
    {
      int mt0 = wave, mt1 = wave + 8;
      v8f acc0 = {}, acc1 = {};
      #pragma unroll
      for (int kc = 0; kc < 5; ++kc) {
        v16bf a = load_tile_A(Qs + nt * 16 * 160 + kc * 32, 160, lane);
        v16bf b0 = load_tile_B(Km + kc * 32 * 240 + mt0 * 16, 240, lane);
        acc0 = wmma_bf16(a, b0, acc0);
        if (mt1 < 15) {
          v16bf b1 = load_tile_B(Km + kc * 32 * 240 + mt1 * 16, 240, lane);
          acc1 = wmma_bf16(a, b1, acc1);
        }
      }
      #pragma unroll
      for (int g = 0; g < 8; ++g)
        attF[(half * 8 + g) * 240 + mt0 * 16 + r] = acc0[g];
      if (mt1 < 15) {
        #pragma unroll
        for (int g = 0; g < 8; ++g)
          attF[(half * 8 + g) * 240 + mt1 * 16 + r] = acc1[g];
      }
    }
    __syncthreads();

    // parallel softmax: 240 threads = 16 rows x 15 segments of 15 columns
    if (tid < 240) {                                   // local max
      const float* ar = attF + srow * 240 + sseg * 15;
      float mx = ar[0];
      #pragma unroll
      for (int j = 1; j < 15; ++j) mx = fmaxf(mx, ar[j]);
      redS[srow * 16 + sseg] = mx;
    }
    __syncthreads();
    if (tid < 16) {                                    // row max
      float mx = redS[tid * 16];
      #pragma unroll
      for (int j = 1; j < 15; ++j) mx = fmaxf(mx, redS[tid * 16 + j]);
      redS[tid * 16 + 15] = mx;
    }
    __syncthreads();
    if (tid < 240) {                                   // exp + local sum
      float mx = redS[srow * 16 + 15];
      float* ar = attF + srow * 240 + sseg * 15;
      float s = 0.0f;
      #pragma unroll
      for (int j = 0; j < 15; ++j) {
        float p = exp2f((ar[j] - mx) * sm_scale);
        ar[j] = p;
        s += p;
      }
      redS[srow * 16 + sseg] = s;
    }
    __syncthreads();
    if (tid < 16) {                                    // row sum -> 1/sum
      float s = redS[tid * 16];
      #pragma unroll
      for (int j = 1; j < 15; ++j) s += redS[tid * 16 + j];
      redS[tid * 16 + 15] = 1.0f / s;
    }
    __syncthreads();
    if (tid < 240) {                                   // normalize, transpose
      float inv = redS[srow * 16 + 15];
      const float* ar = attF + srow * 240 + sseg * 15;
      #pragma unroll
      for (int j = 0; j < 15; ++j)
        attT[(sseg * 15 + j) * 16 + srow] = (bf16_t)(ar[j] * inv);
    }
    __syncthreads();

    // Y tile (256c x 16n) = Xs(c,m) * attT(m,n) ; 2 c-tiles per wave
    {
      int ct0 = wave, ct1 = wave + 8;
      v8f acc0 = {}, acc1 = {};
      #pragma unroll
      for (int kc = 0; kc < 8; ++kc) {
        v16bf b = load_tile_B(attT + kc * 32 * 16, 16, lane);
        v16bf a0 = load_tile_A(Xs + ct0 * 16 * 256 + kc * 32, 256, lane);
        acc0 = wmma_bf16(a0, b, acc0);
        v16bf a1 = load_tile_A(Xs + ct1 * 16 * 256 + kc * 32, 256, lane);
        acc1 = wmma_bf16(a1, b, acc1);
      }
      #pragma unroll
      for (int g = 0; g < 8; ++g) {
        yB[(ct0 * 16 + half * 8 + g) * 16 + r] = (bf16_t)acc0[g];
        yB[(ct1 * 16 + half * 8 + g) * 16 + r] = (bf16_t)acc1[g];
      }
    }
    __syncthreads();

    // out tile (256o x 16n) = Wproj(o,c) * yB(c,n), + residual x, store fp32
    {
      int ot0 = wave, ot1 = wave + 8;
      v8f acc0 = {}, acc1 = {};
      #pragma unroll
      for (int kc = 0; kc < 8; ++kc) {
        v16bf b = load_tile_B(yB + kc * 32 * 16, 16, lane);
        v16bf a0 = load_tile_A(wpb + (size_t)ot0 * 16 * 256 + kc * 32, 256, lane);
        acc0 = wmma_bf16(a0, b, acc0);
        v16bf a1 = load_tile_A(wpb + (size_t)ot1 * 16 * 256 + kc * 32, 256, lane);
        acc1 = wmma_bf16(a1, b, acc1);
      }
      int n = nt * 16 + r;
      if (n < 225) {
        int pr = n / 15, pc = n % 15;
        size_t base = ((size_t)bi * 256) * 57600 + (size_t)(h0 + pr) * 240 + (w0 + pc);
        #pragma unroll
        for (int g = 0; g < 8; ++g) {
          size_t ga0 = base + (size_t)(ot0 * 16 + half * 8 + g) * 57600;
          out[ga0] = x[ga0] + acc0[g];
          size_t ga1 = base + (size_t)(ot1 * 16 + half * 8 + g) * 57600;
          out[ga1] = x[ga1] + acc1[g];
        }
      }
    }
    __syncthreads();
  }
}

// ---- launcher --------------------------------------------------------------
extern "C" void kernel_launch(void* const* d_in, const int* in_sizes, int n_in,
                              void* d_out, int out_size, void* d_ws, size_t ws_size,
                              hipStream_t stream) {
  (void)in_sizes; (void)n_in; (void)out_size; (void)ws_size;
  const float* x  = (const float*)d_in[0];
  const float* wq = (const float*)d_in[1];
  const float* wk = (const float*)d_in[2];
  const float* wp = (const float*)d_in[3];
  bf16_t* wsb = (bf16_t*)d_ws;

  pnlm_prep_weights<<<576, 256, 0, stream>>>(wq, wk, wp, wsb);

  hipFuncSetAttribute((const void*)pnlm_main,
                      hipFuncAttributeMaxDynamicSharedMemorySize, LDS_BYTES);
  pnlm_main<<<2048, 256, LDS_BYTES, stream>>>(x, wsb, wsb + 40960, wsb + 81920,
                                              (float*)d_out);
}
